// VisionTransformer_30193620091383
// MI455X (gfx1250) — matrix operand order repaired and checked by reading, not compiled, not run on hardware
//
#include <hip/hip_runtime.h>

// ---------------------------------------------------------------------------
// BEiT/ViT-Base forward for MI455X (gfx1250).
//  - bf16 WMMA GEMMs: 128x128 block tile, 8 WMMA/wave/K-step, double-buffered
//    LDS staged by the Tensor Data Mover (tensor_load_to_lds + s_wait_tensorcnt)
//  - fully fused attention (scores + rel-pos bias + softmax + P.V) per
//    (batch*head, 64-query tile) block with register-resident softmax
// ---------------------------------------------------------------------------

typedef unsigned short ushort_t;
typedef __attribute__((ext_vector_type(16))) __bf16    bf16x16;
typedef __attribute__((ext_vector_type(8)))  float     f32x8;
typedef __attribute__((ext_vector_type(4)))  unsigned  uint4v;
typedef __attribute__((ext_vector_type(8)))  int       int8v;
typedef __attribute__((ext_vector_type(4)))  int       int4v;

#define DEVI __device__ __forceinline__

#if __has_builtin(__builtin_amdgcn_tensor_load_to_lds)
#define VIT_USE_TDM 1
#else
#define VIT_USE_TDM 0
#endif

namespace vit {
constexpr int B      = 32;
constexpr int IMG    = 224;
constexpr int CIN    = 3;
constexpr int GRID   = 14;
constexpr int NPATCH = 196;
constexpr int NTOK   = 197;
constexpr int D      = 768;
constexpr int DEPTH  = 12;
constexpr int H      = 12;
constexpr int HD     = 64;
constexpr int MLP    = 3072;
constexpr int NCLS   = 1000;
constexpr int NRD    = 27 * 27 + 3;  // 732
constexpr int NPAD   = 224;          // padded token count (7*32)
constexpr int M_TOK  = B * NTOK;     // 6304
}

// ---- scalar helpers --------------------------------------------------------

DEVI ushort_t f2bf(float f) {           // f32 -> bf16 RNE
  unsigned u = __float_as_uint(f);
  unsigned r = u + 0x7FFFu + ((u >> 16) & 1u);
  return (ushort_t)(r >> 16);
}

DEVI float gelu_exact(float x) {
  return 0.5f * x * (1.0f + erff(x * 0.70710678118654752440f));
}

DEVI f32x8 wmma_bf16(bf16x16 a, bf16x16 b, f32x8 c) {
  return __builtin_amdgcn_wmma_f32_16x16x32_bf16(false, a, false, b,
                                                 (short)0, c, false, false);
}

// A fragment (16x32 bf16): lanes 0-15 hold K0-15; lanes 16-31 K16-31.
DEVI bf16x16 load_a(const ushort_t* lds, int row0, int ldk, int lane) {
  const ushort_t* p = lds + (row0 + (lane & 15)) * ldk + ((lane >> 4) * 16);
  bf16x16 a;
#pragma unroll
  for (int i = 0; i < 8; ++i) a[i] = __builtin_bit_cast(__bf16, p[i]);
#pragma unroll
  for (int i = 0; i < 8; ++i) a[8 + i] = __builtin_bit_cast(__bf16, p[8 + i]);
  return a;
}

// B fragment (32x16 bf16) from LDS tile stored [n][k]: lanes 0-15 K0-15, 16-31 K16-31.
DEVI bf16x16 load_b(const ushort_t* lds, int col0, int ldk, int lane) {
  const ushort_t* p = lds + (col0 + (lane & 15)) * ldk + ((lane >> 4) * 16);
  bf16x16 b;
#pragma unroll
  for (int i = 0; i < 16; ++i) b[i] = __builtin_bit_cast(__bf16, p[i]);
  return b;
}

DEVI int rel_idx(int q, int k) {
  if (q == 0 && k == 0) return vit::NRD - 1;
  if (k == 0)           return vit::NRD - 2;
  if (q == 0)           return vit::NRD - 3;
  int qi = q - 1, ki = k - 1;
  int dy = qi / vit::GRID - ki / vit::GRID + (vit::GRID - 1);
  int dx = qi % vit::GRID - ki % vit::GRID + (vit::GRID - 1);
  return dy * (2 * vit::GRID - 1) + dx;
}

// ---- TDM staging: copy a [rows<=128 x 32] bf16 tile into LDS ---------------
// LDS rows padded to 40 ushorts (TDM pad: 4 dwords after every 16 dwords).
// Row OOB (rows < 128) is zero-filled by the TDM itself.

DEVI void stage_tile(ushort_t (*dst)[40], const ushort_t* src, int lda,
                     int rows, int tid) {
#if VIT_USE_TDM
  if (tid < 32) {                       // wave 0 issues one TDM op
    const unsigned long long ga = (unsigned long long)src;
    const unsigned lds_off = (unsigned)(unsigned long long)&dst[0][0];
    uint4v g0;
    g0.x = 1u;                                      // count=1, user mode
    g0.y = lds_off;                                 // lds_addr
    g0.z = (unsigned)(ga & 0xffffffffu);            // global_addr[31:0]
    g0.w = (unsigned)((ga >> 32) & 0x01ffffffu) | (2u << 30);  // [56:32] | type=2
    const unsigned rows_u = (unsigned)rows;
    int8v g1;
    g1[0] = (int)((1u << 16) | (1u << 20) | (3u << 22) | (3u << 25));
    g1[1] = (int)(32u << 16);                       // tensor_dim0[15:0]=32
    g1[2] = (int)((rows_u & 0xffffu) << 16);        // dim0 hi=0 | dim1 lo
    g1[3] = (int)(((rows_u >> 16) & 0xffffu) | (32u << 16));  // dim1 hi | tile0
    g1[4] = (int)128u;                              // tile_dim1 (tile2=0)
    g1[5] = (int)(unsigned)lda;                     // dim0_stride[31:0]
    g1[6] = 0;
    g1[7] = 0;
    int4v z4 = {0, 0, 0, 0};
#if __clang_major__ >= 23
    int8v z8 = {0, 0, 0, 0, 0, 0, 0, 0};
    __builtin_amdgcn_tensor_load_to_lds(g0, g1, z4, z4, z8, 0);
#else
    __builtin_amdgcn_tensor_load_to_lds(g0, g1, z4, z4, 0);
#endif
  }
#else
  const int r = tid >> 1, c0 = (tid & 1) * 16;
  const ushort_t* s = src + (size_t)r * lda + c0;
#pragma unroll
  for (int j = 0; j < 16; ++j) dst[r][c0 + j] = (r < rows) ? s[j] : (ushort_t)0;
#endif
}

DEVI void stage_wait(int tid) {
#if VIT_USE_TDM
  if (tid < 32) __builtin_amdgcn_s_wait_tensorcnt(0);
#endif
}

// ---- bf16 WMMA GEMM: C[M,N] = epi(A[M,K] @ W[N,K]^T + bias) ----------------
// EPI 0: +bias. EPI 1: gelu(+bias). EPI 2: resid + gamma*(+bias).
// OBF: write bf16 output instead of f32.  Requires K % 32 == 0.

template <int EPI, bool OBF>
__global__ __launch_bounds__(256) void gemm_bf16(
    const ushort_t* __restrict__ A, int lda, const ushort_t* __restrict__ W,
    const float* __restrict__ bias, void* __restrict__ Cv, int ldc,
    int M, int N, int K,
    const float* __restrict__ resid, const float* __restrict__ gamma) {
  __shared__ ushort_t As[2][128][40];
  __shared__ ushort_t Ws[2][128][40];

  const int tid = threadIdx.x, lane = tid & 31, wave = tid >> 5;
  const int wm = wave & 3, wn = wave >> 2;
  const int n0 = blockIdx.x * 128, m0 = blockIdx.y * 128;
  const int rowsA = M - m0, rowsW = N - n0;

  f32x8 acc[8];
#pragma unroll
  for (int i = 0; i < 8; ++i) acc[i] = (f32x8){};

  stage_tile(As[0], A + (size_t)m0 * lda, lda, rowsA, tid);
  stage_tile(Ws[0], W + (size_t)n0 * K, K, rowsW, tid);
  stage_wait(tid);
  __syncthreads();

  const int KT = K >> 5;
  for (int kt = 0; kt < KT; ++kt) {
    const int cur = kt & 1, nxt = cur ^ 1;
    if (kt + 1 < KT) {
      stage_tile(As[nxt], A + (size_t)m0 * lda + (kt + 1) * 32, lda, rowsA, tid);
      stage_tile(Ws[nxt], W + (size_t)n0 * K + (kt + 1) * 32, K, rowsW, tid);
    }
    bf16x16 a0 = load_a(&As[cur][0][0], wm * 32, 40, lane);
    bf16x16 a1 = load_a(&As[cur][0][0], wm * 32 + 16, 40, lane);
    bf16x16 b0 = load_b(&Ws[cur][0][0], wn * 64, 40, lane);
    bf16x16 b1 = load_b(&Ws[cur][0][0], wn * 64 + 16, 40, lane);
    bf16x16 b2 = load_b(&Ws[cur][0][0], wn * 64 + 32, 40, lane);
    bf16x16 b3 = load_b(&Ws[cur][0][0], wn * 64 + 48, 40, lane);
    acc[0] = wmma_bf16(a0, b0, acc[0]);
    acc[1] = wmma_bf16(a0, b1, acc[1]);
    acc[2] = wmma_bf16(a0, b2, acc[2]);
    acc[3] = wmma_bf16(a0, b3, acc[3]);
    acc[4] = wmma_bf16(a1, b0, acc[4]);
    acc[5] = wmma_bf16(a1, b1, acc[5]);
    acc[6] = wmma_bf16(a1, b2, acc[6]);
    acc[7] = wmma_bf16(a1, b3, acc[7]);
    if (kt + 1 < KT) {
      stage_wait(tid);
      __syncthreads();
    }
  }

  const int mb = m0 + wm * 32 + (lane >> 4) * 8;
  const int nb = n0 + wn * 64 + (lane & 15);
#pragma unroll
  for (int ai = 0; ai < 2; ++ai) {
#pragma unroll
    for (int bi = 0; bi < 4; ++bi) {
      const f32x8 av = acc[ai * 4 + bi];
      const int n = nb + bi * 16;
      if (n >= N) continue;
#pragma unroll
      for (int j = 0; j < 8; ++j) {
        const int m = mb + ai * 16 + j;
        if (m >= M) continue;
        float v = av[j] + (bias ? bias[n] : 0.0f);
        if (EPI == 1) v = gelu_exact(v);
        if (EPI == 2) v = resid[(size_t)m * ldc + n] + gamma[n] * v;
        if (OBF) ((ushort_t*)Cv)[(size_t)m * ldc + n] = f2bf(v);
        else     ((float*)Cv)[(size_t)m * ldc + n] = v;
      }
    }
  }
}

// ---- fused attention -------------------------------------------------------
// grid (qTiles=4, B*H).  One block: 64 query rows, full 224-padded key range.
// LDS aliasing (static 61.5 KB):
//   region1 (32256B): K tile 224x72  ->  P tile 64x232 (after scores done)
//   region2 (29696B): Q tile 64x72   ->  V tile 64x232 (after scores done)
//   red    (1024B):   per-row partial max / sum across the k-split waves

__global__ __launch_bounds__(256) void attn_fused(
    const float* __restrict__ qkv, const float* __restrict__ biasx,
    ushort_t* __restrict__ O, float scale) {
  using namespace vit;
  __shared__ __align__(16) char smem[32256 + 29696 + 1024];
  ushort_t (*Ks)[72]   = (ushort_t(*)[72])smem;             // 224 x 72
  ushort_t (*Pbf)[232] = (ushort_t(*)[232])smem;            // 64 x 232 (alias)
  ushort_t (*Qs)[72]   = (ushort_t(*)[72])(smem + 32256);   // 64 x 72
  ushort_t (*Vs)[232]  = (ushort_t(*)[232])(smem + 32256);  // 64 x 232 (alias)
  float* pmax = (float*)(smem + 32256 + 29696);             // [64][2]
  float* psum = pmax + 128;                                 // [64][2]

  const int bh = blockIdx.y;
  const int b = bh / H, h = bh % H;
  const int q0 = blockIdx.x * 64;
  const int tid = threadIdx.x, lane = tid & 31, wave = tid >> 5;
  const int wm = wave & 3;        // q sub-tile (16 rows)
  const int kh = wave >> 2;       // k half (112 cols) for scores
  const int hi = lane >> 4;       // lane half
  const int ln = lane & 15;

  // ---- stage Q (64x64) and K (224x64), bf16, zero-padded ----
  {
    const int r = tid >> 2, c0 = (tid & 3) * 16;
    const int gq = q0 + r;
    const float* srcq = qkv + (size_t)(b * NTOK + gq) * (3 * D) + h * HD;
#pragma unroll
    for (int j = 0; j < 16; ++j)
      Qs[r][c0 + j] = f2bf((gq < NTOK) ? srcq[c0 + j] : 0.0f);
  }
  for (int i = tid; i < 224 * 4; i += 256) {
    const int r = i >> 2, c0 = (i & 3) * 16;
    const float* srck = qkv + (size_t)(b * NTOK + r) * (3 * D) + D + h * HD;
#pragma unroll
    for (int j = 0; j < 16; ++j)
      Ks[r][c0 + j] = f2bf((r < NTOK) ? srck[c0 + j] : 0.0f);
  }
  __syncthreads();                              // B1

  // ---- scores: rows [wm*16,+16), cols [kh*112,+112), Kdim=64 ----
  f32x8 acc[7];
#pragma unroll
  for (int i = 0; i < 7; ++i) acc[i] = (f32x8){};
#pragma unroll
  for (int kc = 0; kc < 2; ++kc) {
    bf16x16 a = load_a(&Qs[0][kc * 32], wm * 16, 72, lane);
#pragma unroll
    for (int nt = 0; nt < 7; ++nt) {
      bf16x16 bk = load_b(&Ks[0][kc * 32], kh * 112 + nt * 16, 72, lane);
      acc[nt] = wmma_bf16(a, bk, acc[nt]);
    }
  }

  // scale + expanded bias (biasx pads k>=197 with -inf); per-row partial max
  float mrow[8];
#pragma unroll
  for (int j = 0; j < 8; ++j) mrow[j] = -3.0e38f;
#pragma unroll
  for (int nt = 0; nt < 7; ++nt) {
#pragma unroll
    for (int j = 0; j < 8; ++j) {
      const int q = q0 + wm * 16 + hi * 8 + j;
      const int qc = (q < NTOK) ? q : (NTOK - 1);
      const int k = kh * 112 + nt * 16 + ln;
      const float v = acc[nt][j] * scale +
                      biasx[((size_t)h * NTOK + qc) * NPAD + k];
      acc[nt][j] = v;
      mrow[j] = fmaxf(mrow[j], v);
    }
  }
#pragma unroll
  for (int j = 0; j < 8; ++j)
#pragma unroll
    for (int m = 1; m < 16; m <<= 1)
      mrow[j] = fmaxf(mrow[j], __shfl_xor(mrow[j], m, 32));
  if (ln == 0) {
#pragma unroll
    for (int j = 0; j < 8; ++j)
      pmax[(wm * 16 + hi * 8 + j) * 2 + kh] = mrow[j];
  }
  __syncthreads();                              // B2

  float gmax[8], srow[8];
#pragma unroll
  for (int j = 0; j < 8; ++j) {
    const int r = wm * 16 + hi * 8 + j;
    gmax[j] = fmaxf(pmax[r * 2], pmax[r * 2 + 1]);
    srow[j] = 0.0f;
  }
#pragma unroll
  for (int nt = 0; nt < 7; ++nt) {
#pragma unroll
    for (int j = 0; j < 8; ++j) {
      const float e = __expf(acc[nt][j] - gmax[j]);
      acc[nt][j] = e;
      srow[j] += e;
    }
  }
#pragma unroll
  for (int j = 0; j < 8; ++j)
#pragma unroll
    for (int m = 1; m < 16; m <<= 1) srow[j] += __shfl_xor(srow[j], m, 32);
  if (ln == 0) {
#pragma unroll
    for (int j = 0; j < 8; ++j)
      psum[(wm * 16 + hi * 8 + j) * 2 + kh] = srow[j];
  }

  // stage V (aliases dead Q region): Vs[d][k] = V[k][d], 64x224
  for (int i = tid; i < 64 * 14; i += 256) {
    const int d = i & 63, c0 = (i >> 6) * 16;
#pragma unroll
    for (int j = 0; j < 16; ++j) {
      const int gk = c0 + j;
      float v = 0.0f;
      if (gk < NTOK)
        v = qkv[(size_t)(b * NTOK + gk) * (3 * D) + 2 * D + h * HD + d];
      Vs[d][c0 + j] = f2bf(v);
    }
  }
  __syncthreads();                              // B3

  // normalize, write P bf16 into LDS (aliases dead K region)
  float inv[8];
#pragma unroll
  for (int j = 0; j < 8; ++j) {
    const int r = wm * 16 + hi * 8 + j;
    inv[j] = 1.0f / (psum[r * 2] + psum[r * 2 + 1]);
  }
#pragma unroll
  for (int nt = 0; nt < 7; ++nt) {
#pragma unroll
    for (int j = 0; j < 8; ++j) {
      const int r = wm * 16 + hi * 8 + j;
      const int k = kh * 112 + nt * 16 + ln;
      Pbf[r][k] = f2bf(acc[nt][j] * inv[j]);
    }
  }
  __syncthreads();                              // B4

  // ---- P.V: rows [wm*16,+16), d cols [dh*32,+32), K loop 7x32 ----
  const int dh = wave >> 2;
  f32x8 o0 = {}, o1 = {};
#pragma unroll
  for (int kt = 0; kt < 7; ++kt) {
    bf16x16 a  = load_a(&Pbf[0][kt * 32], wm * 16, 232, lane);
    bf16x16 b0 = load_b(&Vs[0][kt * 32], dh * 32, 232, lane);
    bf16x16 b1 = load_b(&Vs[0][kt * 32], dh * 32 + 16, 232, lane);
    o0 = wmma_bf16(a, b0, o0);
    o1 = wmma_bf16(a, b1, o1);
  }
#pragma unroll
  for (int j = 0; j < 8; ++j) {
    const int q = q0 + wm * 16 + hi * 8 + j;
    if (q >= NTOK) continue;
    ushort_t* dst = O + (size_t)(b * NTOK + q) * D + h * HD;
    dst[dh * 32 + ln]      = f2bf(o0[j]);
    dst[dh * 32 + 16 + ln] = f2bf(o1[j]);
  }
}

// ---- elementwise / reduction kernels ---------------------------------------

__global__ __launch_bounds__(256) void layernorm_bf(
    const float* __restrict__ x, const float* __restrict__ s,
    const float* __restrict__ bb, ushort_t* __restrict__ y) {
  using namespace vit;
  const size_t row = blockIdx.x;
  const int tid = threadIdx.x;
  __shared__ float red[256];
  const float* xr = x + row * D;

  const float v0 = xr[tid], v1 = xr[tid + 256], v2 = xr[tid + 512];
  red[tid] = v0 + v1 + v2;
  __syncthreads();
  for (int st = 128; st > 0; st >>= 1) {
    if (tid < st) red[tid] += red[tid + st];
    __syncthreads();
  }
  const float mean = red[0] * (1.0f / D);
  __syncthreads();
  const float d0 = v0 - mean, d1 = v1 - mean, d2 = v2 - mean;
  red[tid] = d0 * d0 + d1 * d1 + d2 * d2;
  __syncthreads();
  for (int st = 128; st > 0; st >>= 1) {
    if (tid < st) red[tid] += red[tid + st];
    __syncthreads();
  }
  const float rstd = rsqrtf(red[0] * (1.0f / D) + 1e-5f);

  ushort_t* yr = y + row * D;
  yr[tid]       = f2bf(d0 * rstd * s[tid]       + bb[tid]);
  yr[tid + 256] = f2bf(d1 * rstd * s[tid + 256] + bb[tid + 256]);
  yr[tid + 512] = f2bf(d2 * rstd * s[tid + 512] + bb[tid + 512]);
}

__global__ __launch_bounds__(256) void cvt_bf16(
    const float* __restrict__ src, ushort_t* __restrict__ dst, long n) {
  long i = (long)blockIdx.x * 256 + threadIdx.x;
  const long stride = (long)gridDim.x * 256;
  for (; i < n; i += stride) dst[i] = f2bf(src[i]);
}

__global__ __launch_bounds__(256) void patchify_bf(
    const float* __restrict__ x, ushort_t* __restrict__ patches) {
  using namespace vit;
  const int bp = blockIdx.x;
  const int b = bp / NPATCH, p = bp % NPATCH;
  const int gy = p / GRID, gx = p % GRID;
  for (int col = threadIdx.x; col < D; col += 256) {
    const int c = col / 256, rem = col % 256;
    const int py = rem / 16, px = rem % 16;
    const size_t src =
        (((size_t)b * CIN + c) * IMG + (gy * 16 + py)) * IMG + gx * 16 + px;
    patches[(size_t)bp * D + col] = f2bf(x[src]);
  }
}

__global__ __launch_bounds__(256) void assemble_tokens(
    const float* __restrict__ pe, const float* __restrict__ cls,
    const float* __restrict__ pos, float* __restrict__ t) {
  using namespace vit;
  const int bn = blockIdx.x;
  const int b = bn / NTOK, n = bn % NTOK;
  float* dst = t + (size_t)bn * D;
  const float* ppos = pos + (size_t)n * D;
  for (int c = threadIdx.x; c < D; c += 256) {
    const float v =
        (n == 0) ? cls[c] : pe[((size_t)b * NPATCH + (n - 1)) * D + c];
    dst[c] = v + ppos[c];
  }
}

__global__ __launch_bounds__(256) void build_qkv_bias(
    const float* __restrict__ qb, const float* __restrict__ vb,
    float* __restrict__ out) {
  using namespace vit;
  const int c = blockIdx.x * 256 + threadIdx.x;
  if (c >= 3 * D) return;
  float v = 0.0f;
  if (c < D)           v = qb[c];
  else if (c >= 2 * D) v = vb[c - 2 * D];
  out[c] = v;
}

__global__ __launch_bounds__(256) void build_relmap(int* __restrict__ map) {
  using namespace vit;
  const int i = blockIdx.x * 256 + threadIdx.x;
  if (i >= NTOK * NPAD) return;
  const int q = i / NPAD, k = i % NPAD;
  map[i] = rel_idx(q, (k < NTOK) ? k : 0);
}

// expand per-layer rel-pos bias into dense [H][NTOK][NPAD]; pad cols = -inf
__global__ __launch_bounds__(256) void bias_expand(
    const float* __restrict__ rpb, const int* __restrict__ relmap,
    float* __restrict__ bx) {
  using namespace vit;
  const int i = blockIdx.x * 256 + threadIdx.x;
  if (i >= H * NTOK * NPAD) return;
  const int h = i / (NTOK * NPAD);
  const int r = i % (NTOK * NPAD);
  const int k = r % NPAD;
  bx[i] = (k < NTOK) ? rpb[relmap[r] * H + h] : -3.0e38f;
}

// ---------------------------------------------------------------------------

extern "C" void kernel_launch(void* const* d_in, const int* in_sizes, int n_in,
                              void* d_out, int out_size, void* d_ws,
                              size_t ws_size, hipStream_t stream) {
  using namespace vit;
  (void)in_sizes; (void)n_in; (void)out_size; (void)ws_size;

  const float* x         = (const float*)d_in[0];
  const float* patch_w   = (const float*)d_in[1];
  const float* patch_b   = (const float*)d_in[2];
  const float* cls_token = (const float*)d_in[3];
  const float* pos_embed = (const float*)d_in[4];
  const float* norm1_s   = (const float*)d_in[5];
  const float* norm1_b   = (const float*)d_in[6];
  const float* qkv_w     = (const float*)d_in[7];
  const float* q_bias    = (const float*)d_in[8];
  const float* v_bias    = (const float*)d_in[9];
  const float* rpb_table = (const float*)d_in[10];
  const float* proj_w    = (const float*)d_in[11];
  const float* proj_b    = (const float*)d_in[12];
  const float* norm2_s   = (const float*)d_in[13];
  const float* norm2_b   = (const float*)d_in[14];
  const float* fc1_w     = (const float*)d_in[15];
  const float* fc1_b     = (const float*)d_in[16];
  const float* fc2_w     = (const float*)d_in[17];
  const float* fc2_b     = (const float*)d_in[18];
  const float* gamma1    = (const float*)d_in[19];
  const float* gamma2    = (const float*)d_in[20];
  const float* normf_s   = (const float*)d_in[21];
  const float* normf_b   = (const float*)d_in[22];
  const float* head_w    = (const float*)d_in[23];
  const float* head_b    = (const float*)d_in[24];
  float* out = (float*)d_out;

  char* wp = (char*)d_ws;
  auto alloc = [&](size_t bytes) {
    char* r = wp;
    wp += (bytes + 255) & ~(size_t)255;
    return r;
  };
  // bf16 weight copies (converted once per launch)
  ushort_t* wb_patch = (ushort_t*)alloc((size_t)D * D * 2);
  ushort_t* wb_qkv   = (ushort_t*)alloc((size_t)DEPTH * 3 * D * D * 2);
  ushort_t* wb_proj  = (ushort_t*)alloc((size_t)DEPTH * D * D * 2);
  ushort_t* wb_fc1   = (ushort_t*)alloc((size_t)DEPTH * MLP * D * 2);
  ushort_t* wb_fc2   = (ushort_t*)alloc((size_t)DEPTH * D * MLP * 2);
  ushort_t* wb_head  = (ushort_t*)alloc((size_t)NCLS * D * 2);
  // activations
  ushort_t* pbf   = (ushort_t*)alloc((size_t)B * NPATCH * D * 2);
  float*    pe    = (float*)alloc((size_t)B * NPATCH * D * 4);
  float*    t     = (float*)alloc((size_t)M_TOK * D * 4);
  ushort_t* hbf   = (ushort_t*)alloc((size_t)M_TOK * D * 2);
  float*    qkvb  = (float*)alloc((size_t)3 * D * 4);
  float*    qkv   = (float*)alloc((size_t)M_TOK * 3 * D * 4);
  ushort_t* ob    = (ushort_t*)alloc((size_t)M_TOK * D * 2);
  ushort_t* m1b   = (ushort_t*)alloc((size_t)M_TOK * MLP * 2);
  int*      rmap  = (int*)alloc((size_t)NTOK * NPAD * 4);
  float*    bx    = (float*)alloc((size_t)H * NTOK * NPAD * 4);

  const dim3 blk(256);
  const float scale = 0.125f;                 // HD^-0.5
  const int mT = (M_TOK + 127) / 128;         // 50

  // one-time per launch: bf16 weight conversion + rel-pos index map
  cvt_bf16<<<1024, blk, 0, stream>>>(patch_w, wb_patch, (long)D * D);
  cvt_bf16<<<2048, blk, 0, stream>>>(qkv_w,  wb_qkv,  (long)DEPTH * 3 * D * D);
  cvt_bf16<<<2048, blk, 0, stream>>>(proj_w, wb_proj, (long)DEPTH * D * D);
  cvt_bf16<<<2048, blk, 0, stream>>>(fc1_w,  wb_fc1,  (long)DEPTH * MLP * D);
  cvt_bf16<<<2048, blk, 0, stream>>>(fc2_w,  wb_fc2,  (long)DEPTH * D * MLP);
  cvt_bf16<<<1024, blk, 0, stream>>>(head_w, wb_head, (long)NCLS * D);
  build_relmap<<<(NTOK * NPAD + 255) / 256, blk, 0, stream>>>(rmap);

  // patch embed
  patchify_bf<<<B * NPATCH, blk, 0, stream>>>(x, pbf);
  gemm_bf16<0, false><<<dim3(D / 128, (B * NPATCH + 127) / 128), blk, 0, stream>>>(
      pbf, D, wb_patch, patch_b, pe, D, B * NPATCH, D, D, nullptr, nullptr);
  assemble_tokens<<<M_TOK, blk, 0, stream>>>(pe, cls_token, pos_embed, t);

  for (int i = 0; i < DEPTH; ++i) {
    layernorm_bf<<<M_TOK, blk, 0, stream>>>(t, norm1_s + i * D, norm1_b + i * D, hbf);
    build_qkv_bias<<<(3 * D + 255) / 256, blk, 0, stream>>>(
        q_bias + i * D, v_bias + i * D, qkvb);
    gemm_bf16<0, false><<<dim3(3 * D / 128, mT), blk, 0, stream>>>(
        hbf, D, wb_qkv + (size_t)i * 3 * D * D, qkvb, qkv, 3 * D,
        M_TOK, 3 * D, D, nullptr, nullptr);

    bias_expand<<<(H * NTOK * NPAD + 255) / 256, blk, 0, stream>>>(
        rpb_table + (size_t)i * NRD * H, rmap, bx);
    attn_fused<<<dim3(4, B * H), blk, 0, stream>>>(qkv, bx, ob, scale);

    gemm_bf16<2, false><<<dim3(D / 128, mT), blk, 0, stream>>>(
        ob, D, wb_proj + (size_t)i * D * D, proj_b + i * D, t, D,
        M_TOK, D, D, t, gamma1 + i * D);

    layernorm_bf<<<M_TOK, blk, 0, stream>>>(t, norm2_s + i * D, norm2_b + i * D, hbf);
    gemm_bf16<1, true><<<dim3(MLP / 128, mT), blk, 0, stream>>>(
        hbf, D, wb_fc1 + (size_t)i * MLP * D, fc1_b + i * MLP, m1b, MLP,
        M_TOK, MLP, D, nullptr, nullptr);
    gemm_bf16<2, false><<<dim3(D / 128, mT), blk, 0, stream>>>(
        m1b, MLP, wb_fc2 + (size_t)i * D * MLP, fc2_b + i * D, t, D,
        M_TOK, D, MLP, t, gamma2 + i * D);
  }

  layernorm_bf<<<M_TOK, blk, 0, stream>>>(t, normf_s, normf_b, hbf);
  gemm_bf16<0, false><<<dim3((NCLS + 127) / 128, 1), blk, 0, stream>>>(
      hbf, NTOK * D, wb_head, head_b, out, NCLS, B, NCLS, D, nullptr, nullptr);
}
